// Colight_51788715655808
// MI455X (gfx1250) — matrix operand order repaired and checked by reading, not compiled, not run on hardware
//
#include <hip/hip_runtime.h>

// Problem constants (match reference)
#define B_N    100000
#define N_NEI  8
#define IN_DIM 64
#define HEADS  5
#define HID    128

typedef __attribute__((ext_vector_type(16))) _Float16 v16h;
typedef __attribute__((ext_vector_type(8)))  float    v8f;

// Packed B-fragment sizes (in halves)
#define PACK_W   (HEADS*8*2*32*16)     // 40960 halves per weight tensor (Wa/Wo/Wf)
#define PACK_FC  (8*4*32*16)           // 16384 halves for W_fc
#define PACK_TOT (3*PACK_W + PACK_FC)  // 139264 halves = 278528 bytes
#define STAGE_H  (16*HID)              // 2048 halves per wave for FC staging
#define WAVES_PB 8
#define LDS_HALVES (PACK_TOT + WAVES_PB*STAGE_H)
#define LDS_BYTES  (LDS_HALVES*2)      // 311296 B <= 320KB WGP LDS
#define NTILES (B_N/16)                // 6250

#define LOG2E 1.4426950408889634f

// -------- prep: repack f32 weights into f16 WMMA B-fragment lane layout --------
// B fragment (16-bit, 32x16 K x N): lane L holds column n = L&15;
//   element j holds K = (L>>4)*8 + j (j<8), 16 + (L>>4)*8 + (j-8) (j>=8) within the 32-wide K slice.
// packed[tensor][h][t][s][L][j], t = 16-col tile, s = 32-wide K slice.
// W_fc is pre-scaled by 1/HEADS:  (final/H) @ W == final @ (W/H).
__global__ void pack_weights_kernel(const float* __restrict__ Wa,
                                    const float* __restrict__ Wo,
                                    const float* __restrict__ Wf,
                                    const float* __restrict__ Wfc,
                                    _Float16* __restrict__ pk) {
  int idx = blockIdx.x * blockDim.x + threadIdx.x;
  if (idx < 3*PACK_W) {
    int m = idx / PACK_W;
    int r = idx % PACK_W;
    int h = r / 8192; r %= 8192;
    int t = r / 1024; r %= 1024;
    int s = r / 512;  r %= 512;
    int L = r / 16;   int j = r % 16;
    int n = t*16 + (L & 15);
    int k = s*32 + (L >> 4)*8 + (j & 7) + (j >> 3)*16;
    const float* W = (m == 0) ? Wa : (m == 1) ? Wo : Wf;
    pk[idx] = (_Float16)W[(h*IN_DIM + k)*HID + n];
  } else if (idx < PACK_TOT) {
    int r = idx - 3*PACK_W;
    int t = r / 2048; r %= 2048;
    int s = r / 512;  r %= 512;
    int L = r / 16;   int j = r % 16;
    int n = t*16 + (L & 15);
    int k = s*32 + (L >> 4)*8 + (j & 7) + (j >> 3)*16;
    pk[idx] = (_Float16)(Wfc[k*HID + n] * (1.0f/HEADS));
  }
}

// -------- fragment helpers --------
// 16-bit A fragment (16x32 M x K): lane L holds row M = L&15;
//   elements j=0..7 -> K = (L>>4)*8 + j ; j=8..15 -> K = 16 + (L>>4)*8 + (j-8)
__device__ __forceinline__ v16h afrag_f32(const float* __restrict__ p0) {
  v16h a;
#pragma unroll
  for (int j = 0; j < 8; ++j) a[j]   = (_Float16)p0[j];
#pragma unroll
  for (int j = 0; j < 8; ++j) a[j+8] = (_Float16)p0[16+j];
  return a;
}
__device__ __forceinline__ v16h afrag_h16(const _Float16* __restrict__ p0) {
  v16h a;
#pragma unroll
  for (int j = 0; j < 8; ++j) a[j]   = p0[j];
#pragma unroll
  for (int j = 0; j < 8; ++j) a[j+8] = p0[16+j];
  return a;
}
__device__ __forceinline__ v16h bfrag_lds(const _Float16* __restrict__ p) {
  v16h b;
#pragma unroll
  for (int j = 0; j < 16; ++j) b[j] = p[j];
  return b;
}
#define WMMA(a, b, c) \
  __builtin_amdgcn_wmma_f32_16x16x32_f16(false, (a), false, (b), (short)0, (c), false, false)

// -------- main fused kernel: one wave per 16-row batch tile --------
__global__ void __launch_bounds__(256, 1) colight_main(
    const float* __restrict__ x, const float* __restrict__ others,
    const int* __restrict__ nm,
    const float* __restrict__ ba, const float* __restrict__ bo,
    const float* __restrict__ bf, const float* __restrict__ bfc,
    const _Float16* __restrict__ pk, float* __restrict__ out) {
  extern __shared__ _Float16 sm[];
  const int tid = threadIdx.x;

  // Stage all packed weights into LDS (16B chunks, whole block cooperates).
  {
    const uint4* src = (const uint4*)pk;
    uint4* dst = (uint4*)sm;
    const int n16 = (PACK_TOT*2)/16;  // 17408 chunks
    for (int i = tid; i < n16; i += 256) dst[i] = src[i];
  }
  __syncthreads();

  const int wave = tid >> 5, lane = tid & 31;
  const int tile = blockIdx.x * WAVES_PB + wave;
  if (tile >= NTILES) return;  // wave-uniform -> EXEC stays all-ones (WMMA req.)

  const int hi = lane >> 4;         // lane half (K-offset group)
  const int ln = lane & 15;         // row (A) / column (B,C,D) index
  const int row = tile*16 + ln;
  _Float16* stage = sm + PACK_TOT + wave*STAGE_H;

  // A fragments for x and all 8 neighbor tiles, kept resident in VGPRs.
  v16h aX[2];
#pragma unroll
  for (int s = 0; s < 2; ++s)
    aX[s] = afrag_f32(x + (size_t)row*IN_DIM + s*32 + hi*8);
  v16h aO[N_NEI][2];
#pragma unroll
  for (int n = 0; n < N_NEI; ++n)
#pragma unroll
    for (int s = 0; s < 2; ++s)
      aO[n][s] = afrag_f32(others + ((size_t)n*B_N + row)*IN_DIM + s*32 + hi*8);

  // Additive mask bias: madd[n][r] = -inf if neighbor n masked for C-layout
  // row m = r + 8*hi, else 0.  exp2(logit + (-inf)) == 0 exactly, so masked
  // slots contribute nothing to the softmax sum or the weighted accumulation.
  v8f madd[N_NEI];
#pragma unroll
  for (int n = 0; n < N_NEI; ++n) {
    const int* mp = nm + (size_t)n*B_N + tile*16 + hi*8;
    int4 a = *(const int4*)mp;
    int4 b = *(const int4*)(mp + 4);
    const float NINF = -__builtin_inff();
    madd[n][0] = (a.x==0) ? NINF : 0.f;
    madd[n][1] = (a.y==0) ? NINF : 0.f;
    madd[n][2] = (a.z==0) ? NINF : 0.f;
    madd[n][3] = (a.w==0) ? NINF : 0.f;
    madd[n][4] = (b.x==0) ? NINF : 0.f;
    madd[n][5] = (b.y==0) ? NINF : 0.f;
    madd[n][6] = (b.z==0) ? NINF : 0.f;
    madd[n][7] = (b.w==0) ? NINF : 0.f;
  }

  const _Float16* pkA = sm;
  const _Float16* pkO = sm + PACK_W;
  const _Float16* pkF = sm + 2*PACK_W;
  const _Float16* pkC = sm + 3*PACK_W;

  for (int t = 0; t < 8; ++t) {           // 16-col output tiles of HID=128
    v8f facc;
#pragma unroll
    for (int r = 0; r < 8; ++r) facc[r] = 0.f;

    for (int h = 0; h < HEADS; ++h) {
      const int fb = (h*8 + t)*1024 + lane*16;  // [h][t][s=0][lane][0]
      v16h bA0 = bfrag_lds(pkA + fb), bA1 = bfrag_lds(pkA + fb + 512);
      v16h bO0 = bfrag_lds(pkO + fb), bO1 = bfrag_lds(pkO + fb + 512);
      v16h bF0 = bfrag_lds(pkF + fb), bF1 = bfrag_lds(pkF + fb + 512);
      float bav = ba[h*HID + t*16 + ln];
      float bov = bo[h*HID + t*16 + ln];
      float bfv = bf[h*HID + t*16 + ln];

      // actor = (x @ Wa[h] + ba[h]) * log2(e)   (exp folded into exp2)
      v8f actor;
#pragma unroll
      for (int r = 0; r < 8; ++r) actor[r] = bav;
      actor = WMMA(aX[0], bA0, actor);
      actor = WMMA(aX[1], bA1, actor);
#pragma unroll
      for (int r = 0; r < 8; ++r) actor[r] *= LOG2E;

      // w[n] = exp2(actor' * (others[n] @ Wo[h] + bo[h]) + madd[n])
      // (no max-shift: |logits| << 128, exp2 cannot overflow f32 here)
      v8f w[N_NEI];
      v8f ssum;
#pragma unroll
      for (int r = 0; r < 8; ++r) ssum[r] = 0.f;
#pragma unroll
      for (int n = 0; n < N_NEI; ++n) {
        v8f o;
#pragma unroll
        for (int r = 0; r < 8; ++r) o[r] = bov;
        o = WMMA(aO[n][0], bO0, o);
        o = WMMA(aO[n][1], bO1, o);
#pragma unroll
        for (int r = 0; r < 8; ++r) {
          float e = __builtin_amdgcn_exp2f(__builtin_fmaf(actor[r], o[r], madd[n][r]));
          w[n][r] = e;
          ssum[r] += e;
        }
      }

      // hacc = sum_n w[n] * (others[n] @ Wf[h] + bf[h]);  facc += hacc / ssum
      v8f hacc;
#pragma unroll
      for (int r = 0; r < 8; ++r) hacc[r] = 0.f;
#pragma unroll
      for (int n = 0; n < N_NEI; ++n) {
        v8f fin;
#pragma unroll
        for (int r = 0; r < 8; ++r) fin[r] = bfv;
        fin = WMMA(aO[n][0], bF0, fin);
        fin = WMMA(aO[n][1], bF1, fin);
#pragma unroll
        for (int r = 0; r < 8; ++r) hacc[r] += w[n][r] * fin[r];
      }
#pragma unroll
      for (int r = 0; r < 8; ++r)
        facc[r] += hacc[r] * __builtin_amdgcn_rcpf(ssum[r]);  // ssum >= 1, mask[0]==1

    }

    // final -> LDS staging (row-major f16 [16][HID]); 1/HEADS is folded into W_fc
#pragma unroll
    for (int r = 0; r < 8; ++r) {
      int m = r + hi*8;  // C layout: VGPR r, lane half hi -> matrix row
      stage[m*HID + t*16 + ln] = (_Float16)facc[r];
    }
  }

  // Make wave's own LDS stores visible to its cross-lane fragment reloads.
  asm volatile("s_wait_dscnt 0" ::: "memory");

  // out = relu(final @ (W_fc/HEADS) + b_fc)   (K=128 -> 4 chained WMMAs per tile)
  v16h aF[4];
#pragma unroll
  for (int s = 0; s < 4; ++s)
    aF[s] = afrag_h16(stage + ln*HID + s*32 + hi*8);

  for (int t = 0; t < 8; ++t) {
    float bcv = bfc[t*16 + ln];
    v8f acc;
#pragma unroll
    for (int r = 0; r < 8; ++r) acc[r] = bcv;
#pragma unroll
    for (int s = 0; s < 4; ++s) {
      v16h bC = bfrag_lds(pkC + ((t*4 + s)*32 + lane)*16);
      acc = WMMA(aF[s], bC, acc);
    }
#pragma unroll
    for (int r = 0; r < 8; ++r) {
      int gr = tile*16 + r + hi*8;
      out[(size_t)gr*HID + t*16 + ln] = fmaxf(acc[r], 0.f);
    }
  }
}

extern "C" void kernel_launch(void* const* d_in, const int* in_sizes, int n_in,
                              void* d_out, int out_size, void* d_ws, size_t ws_size,
                              hipStream_t stream) {
  const float* x      = (const float*)d_in[0];
  const float* others = (const float*)d_in[1];
  const int*   nmask  = (const int*)  d_in[2];
  const float* Wa     = (const float*)d_in[3];
  const float* ba     = (const float*)d_in[4];
  const float* Wo     = (const float*)d_in[5];
  const float* bo     = (const float*)d_in[6];
  const float* Wf     = (const float*)d_in[7];
  const float* bf     = (const float*)d_in[8];
  const float* Wfc    = (const float*)d_in[9];
  const float* bfc    = (const float*)d_in[10];
  float* out          = (float*)d_out;

  _Float16* pk = (_Float16*)d_ws;  // needs 278528 B of workspace

  (void)in_sizes; (void)n_in; (void)out_size; (void)ws_size;

  // 1) repack weights f32 -> f16 WMMA B-fragment layout (W_fc pre-scaled by 1/5)
  pack_weights_kernel<<<(PACK_TOT + 255)/256, 256, 0, stream>>>(Wa, Wo, Wf, Wfc, pk);

  // 2) fused attention + FC; one wave per 16-row tile, 8 waves per block
  const int blocks = (NTILES + WAVES_PB - 1) / WAVES_PB;  // 782
  colight_main<<<blocks, 256, LDS_BYTES, stream>>>(
      x, others, nmask, ba, bo, bf, bfc, pk, out);
}
